// lstm_model_55937654063456
// MI455X (gfx1250) — matrix operand order repaired and checked by reading, not compiled, not run on hardware
//
#include <hip/hip_runtime.h>
#include <stdint.h>

// ---------------- problem constants (match reference) ----------------
constexpr int T   = 1024;
constexpr int BB  = 32;      // batch
constexpr int I   = 128;     // input dim
constexpr int H   = 512;     // hidden dim
constexpr int OUT = 10;
constexpr int G   = 4 * H;   // 2048 gate columns
constexpr int M1  = T * BB;  // 32768 rows for the batched projections

// recurrent persistent-kernel partitioning
constexpr int NWG    = 16;        // workgroups in recurrent kernel
constexpr int HSLICE = H / NWG;   // 32 h-columns per WG
constexpr int GCOLS  = 4 * HSLICE;// 128 gate columns per WG

// ---------------- WMMA vector types ----------------
typedef __attribute__((ext_vector_type(16))) __bf16    v16bf;
typedef __attribute__((ext_vector_type(8)))  float     v8f;
typedef __attribute__((ext_vector_type(8)))  uint32_t  v8u;

// bf16 <-> f32 without scalar __bf16 arithmetic
__device__ inline float bf2f(uint16_t u) {
  uint32_t x = ((uint32_t)u) << 16;
  return __builtin_bit_cast(float, x);
}
__device__ inline uint16_t f2bf(float f) {
  uint32_t x = __builtin_bit_cast(uint32_t, f);
  uint32_t r = x + 0x7FFFu + ((x >> 16) & 1u);   // round-to-nearest-even
  return (uint16_t)(r >> 16);
}

// ---------------- Tensor Data Mover (gfx1250) ----------------
#ifndef HAVE_TDM
#if defined(__has_builtin)
#if __has_builtin(__builtin_amdgcn_tensor_load_to_lds) && __has_builtin(__builtin_amdgcn_s_wait_tensorcnt)
#define HAVE_TDM 1
#endif
#endif
#endif
#ifndef HAVE_TDM
#define HAVE_TDM 0
#endif

#if HAVE_TDM
typedef __attribute__((ext_vector_type(4))) uint32_t v4u;
typedef __attribute__((ext_vector_type(8))) int      v8i;
typedef __attribute__((ext_vector_type(4))) int      v4i;

// Build a D# per CDNA5 ISA and issue TENSOR_LOAD_TO_LDS.
// Strides/dims are in data_size units (data_size_code: 0=1B,1=2B,2=4B,3=8B).
__device__ inline void tdm_load_to_lds(uint32_t lds_addr, uint64_t gaddr,
                                       uint32_t dsz_code,
                                       uint32_t tile0, uint32_t tile1, uint32_t tile2,
                                       uint32_t tdim0, uint32_t tdim1, uint32_t tdim2,
                                       uint64_t stride0, uint64_t stride1) {
  v4u g0;
  g0[0] = 1u;                                        // count=1 (valid), user mode
  g0[1] = lds_addr;                                  // LDS byte address
  g0[2] = (uint32_t)gaddr;                           // global_addr[31:0]
  g0[3] = ((uint32_t)(gaddr >> 32) & 0x01FFFFFFu) | 0x80000000u;  // addr[56:32] | type=2
  v8i g1;
  g1[0] = (int)(dsz_code << 16);                                         // mask=0, data_size
  g1[1] = (int)((tdim0 & 0xFFFFu) << 16);                                // tensor_dim0 lo16
  g1[2] = (int)((tdim0 >> 16) | ((tdim1 & 0xFFFFu) << 16));              // dim0 hi | dim1 lo
  g1[3] = (int)(uint32_t)((tdim1 >> 16) | (tile0 << 16));                // dim1 hi | tile0
  g1[4] = (int)(uint32_t)((tile1 & 0xFFFFu) | (tile2 << 16));            // tile1 | tile2
  g1[5] = (int)(uint32_t)stride0;                                        // dim0_stride lo32
  g1[6] = (int)(uint32_t)(((stride0 >> 32) & 0xFFFFu) |
                          ((uint32_t)(stride1 & 0xFFFFu) << 16));        // d0s hi | d1s lo
  g1[7] = (int)(uint32_t)(stride1 >> 16);                                // dim1_stride hi32
  v4i g2;
  g2[0] = (int)tdim2; g2[1] = 0; g2[2] = 0; g2[3] = 0;                   // tensor_dim2
  v4i g3 = {0, 0, 0, 0};
#if __clang_major__ >= 23
  v8i g4 = {0, 0, 0, 0, 0, 0, 0, 0};
  __builtin_amdgcn_tensor_load_to_lds(g0, g1, g2, g3, g4, 0);
#else
  __builtin_amdgcn_tensor_load_to_lds(g0, g1, g2, g3, 0);
#endif
}
#endif  // HAVE_TDM

// ---------------- WMMA fragment loaders (contiguous k = 16*(lane/16) + s, same in A and B) ----------------
// A fragment: 16x32 bf16 (MxK), row-major source with leading dim lda.
// lanes 0..15 hold K=0..15, lanes 16..31 hold K=16..31 (VGPR v -> K pair 2v).
__device__ inline v16bf load_frag_a(const uint16_t* A, int lda, int row0, int k0) {
  const int lane = threadIdx.x & 31;
  const int m = lane & 15;
  const int g = lane >> 4;
  const uint32_t* p = (const uint32_t*)(A + (size_t)(row0 + m) * lda + k0);
  v8u f;
#pragma unroll
  for (int v = 0; v < 8; ++v) {
    f[v] = p[g * 8 + v];     // dword index = K/2
  }
  return __builtin_bit_cast(v16bf, f);
}

// B fragment: 32x16 bf16 (KxN) where B[k][n] = W[n0+n][k0+k] (W row-major [N,K]).
// lanes 0..15 hold K=0..15, lanes 16..31 hold K=16..31 (VGPR v -> K pair 2v).
__device__ inline v16bf load_frag_b(const uint16_t* W, int ldw, int n0, int k0) {
  const int lane = threadIdx.x & 31;
  const int n = lane & 15;
  const int g = lane >> 4;
  const uint32_t* p = (const uint32_t*)(W + (size_t)(n0 + n) * ldw + k0);
  v8u f;
#pragma unroll
  for (int v = 0; v < 8; ++v) f[v] = p[g * 8 + v];  // contiguous 32B per lane
  return __builtin_bit_cast(v16bf, f);
}

__device__ inline float fast_sigmoid(float x) { return 1.0f / (1.0f + __expf(-x)); }

// ---------------- elementwise f32 -> bf16 convert ----------------
__global__ void f32_to_bf16_k(const float* __restrict__ in, uint16_t* __restrict__ out, int n) {
  for (int i = blockIdx.x * blockDim.x + threadIdx.x; i < n; i += gridDim.x * blockDim.x)
    out[i] = f2bf(in[i]);
}

// ---------------- zero h-staging + sync vars ----------------
__global__ void init_state_k(uint32_t* __restrict__ hstage_dw, int ndw, int* __restrict__ sync) {
  int i = blockIdx.x * blockDim.x + threadIdx.x;
  if (i < ndw) hstage_dw[i] = 0u;
  if (i < 8)   sync[i] = 0;
}

// ---------------- batched input projection: C[M,N] = A[M,K] * W[N,K]^T + bias ----------------
// 256 threads = 8 waves; block tile = 32(M) x 256(N); each wave: 2x2 WMMA tiles.
__global__ void __launch_bounds__(256, 2)
gemm_bias_wmma(const uint16_t* __restrict__ A, const uint16_t* __restrict__ W,
               const float* __restrict__ bias, float* __restrict__ C,
               int M, int N, int K) {
  const int wave = threadIdx.x >> 5;
  const int nbn  = N >> 8;                 // N / 256
  const int bm   = blockIdx.x / nbn;
  const int bn   = blockIdx.x % nbn;
  const int row0 = bm * 32;
  const int col0 = bn * 256 + wave * 32;

  v8f acc00 = {}, acc01 = {}, acc10 = {}, acc11 = {};
  for (int k0 = 0; k0 < K; k0 += 32) {
    v16bf a0 = load_frag_a(A, K, row0,      k0);
    v16bf a1 = load_frag_a(A, K, row0 + 16, k0);
    v16bf b0 = load_frag_b(W, K, col0,      k0);
    v16bf b1 = load_frag_b(W, K, col0 + 16, k0);
    acc00 = __builtin_amdgcn_wmma_f32_16x16x32_bf16(false, a0, false, b0, (short)0, acc00, false, false);
    acc01 = __builtin_amdgcn_wmma_f32_16x16x32_bf16(false, a0, false, b1, (short)0, acc01, false, false);
    acc10 = __builtin_amdgcn_wmma_f32_16x16x32_bf16(false, a1, false, b0, (short)0, acc10, false, false);
    acc11 = __builtin_amdgcn_wmma_f32_16x16x32_bf16(false, a1, false, b1, (short)0, acc11, false, false);
  }

  const int lane = threadIdx.x & 31;
  const int n  = lane & 15;
  const int mo = (lane >> 4) * 8;
  const float bv0 = bias[col0 + n];
  const float bv1 = bias[col0 + 16 + n];
#pragma unroll
  for (int v = 0; v < 8; ++v) {
    const size_t r0 = (size_t)(row0 + mo + v)      * N;
    const size_t r1 = (size_t)(row0 + 16 + mo + v) * N;
    C[r0 + col0 + n]      = acc00[v] + bv0;
    C[r0 + col0 + 16 + n] = acc01[v] + bv1;
    C[r1 + col0 + n]      = acc10[v] + bv0;
    C[r1 + col0 + 16 + n] = acc11[v] + bv1;
  }
}

// ---------------- grid-wide sense barrier (16 co-resident WGs) ----------------
__device__ inline void grid_sync(int* cnt, volatile int* flag, int target) {
  __syncthreads();
  if (threadIdx.x == 0) {
    __threadfence();
    int old = atomicAdd(cnt, 1);
    if (old == NWG - 1) {
      *cnt = 0;
      __threadfence();
      *flag = target;
    } else {
      while (*flag < target) __builtin_amdgcn_s_sleep(2);
      __threadfence();
    }
  }
  __syncthreads();
}

// ---------------- persistent LSTM recurrence ----------------
// 16 WGs, each owns 32 h-columns (=> 128 gate rows). W_hh slice resident in LDS (128 KB),
// staged by the Tensor Data Mover; per-step h tile (32 KB) also DMA'd via TDM.
// Dynamic LDS layout: whh[128*512] bf16 | hsh[32*512] bf16 | gates[32*128] f32 | c[32*32] f32.
__global__ void __launch_bounds__(256, 1)
lstm_recur_wmma(const float* __restrict__ xproj,      // [T,B,4H] f32 (bias included)
                const uint16_t* __restrict__ Whh,     // [4H,H] bf16
                uint16_t* __restrict__ hstage,        // [2,B,H] bf16 (zero-initialized)
                uint16_t* __restrict__ hs_out,        // [T,B,H] bf16
                int* __restrict__ cnt, int* __restrict__ flag) {
  extern __shared__ char smem[];
  uint16_t* whh_sh = (uint16_t*)smem;                                       // 128*512 bf16
  uint16_t* hsh    = (uint16_t*)(smem + (size_t)GCOLS * H * 2);             // 32*512 bf16
  float*    gsh    = (float*)(smem + (size_t)(GCOLS + BB) * H * 2);         // 32*128 f32
  float*    csh    = (float*)((char*)gsh + (size_t)BB * GCOLS * 4);         // 32*32 f32

  const int tid = threadIdx.x;
  const int wg  = blockIdx.x;          // 0..15
  const int hc0 = wg * HSLICE;

  // Preload this WG's W_hh slice: local row l = q*32+j  <->  global row q*H + hc0 + j.
#if HAVE_TDM
  if (tid < 32) {
    // 3-D tile in 8-byte units: x = one 512-elem bf16 row (128 units), y = 32 rows
    // (row stride H*2B = 128 units), z = the 4 gate blocks (stride H rows = H*H/4 units).
    tdm_load_to_lds((uint32_t)(uintptr_t)(void*)whh_sh,
                    (uint64_t)(uintptr_t)(Whh + (size_t)hc0 * H),
                    /*dsz=8B*/ 3,
                    /*tile */ H / 4, HSLICE, 4,
                    /*tensor*/ H / 4, HSLICE, 4,
                    /*stride0*/ (uint64_t)(H / 4),
                    /*stride1*/ (uint64_t)H * H / 4);
    __builtin_amdgcn_s_wait_tensorcnt(0);
  }
#else
  {
    const uint32_t* src = (const uint32_t*)Whh;
    uint32_t* dst = (uint32_t*)whh_sh;
    const int rowdw = H / 2;  // 256 dwords per row
    for (int dw = tid; dw < GCOLS * rowdw; dw += 256) {
      const int l = dw >> 8;
      const int c = dw & 255;
      const int q = l >> 5, j = l & 31;
      dst[(size_t)l * rowdw + c] = src[(size_t)(q * H + hc0 + j) * rowdw + c];
    }
  }
#endif
  for (int i = tid; i < BB * HSLICE; i += 256) csh[i] = 0.0f;
  __syncthreads();

  const int wave  = tid >> 5;
  const int lane  = tid & 31;
  const int q     = wave >> 1;          // which gate (i,f,g,o) this wave's tile lives in
  const int jb    = (wave & 1) * 16;    // offset within the 32-wide gate slice
  const int lrow0 = wave * 16;          // local row in whh_sh

  for (int t = 0; t < T; ++t) {
    // Stage previous h (full [B,H] = 32 KB) from the global double-buffer into LDS.
#if HAVE_TDM
    if (wave == 0) {
      tdm_load_to_lds((uint32_t)(uintptr_t)(void*)hsh,
                      (uint64_t)(uintptr_t)(hstage + (size_t)(t & 1) * BB * H),
                      /*dsz=8B*/ 3,
                      /*tile */ BB * H / 4, 1, 0,
                      /*tensor*/ BB * H / 4, 1, 0,
                      /*stride0*/ (uint64_t)(BB * H / 4),
                      /*stride1*/ 0);
      __builtin_amdgcn_s_wait_tensorcnt(0);
    }
#else
    {
      const uint32_t* src = (const uint32_t*)(hstage + (size_t)(t & 1) * BB * H);
      uint32_t* dst = (uint32_t*)hsh;
      for (int dw = tid; dw < BB * H / 2; dw += 256) dst[dw] = src[dw];
    }
#endif
    __syncthreads();

    // gates_slice[32 x 128] = h_prev[32x512] * Whh_slice^T, per-wave 16 N-cols, 2 M-tiles.
    v8f acc0 = {}, acc1 = {};
#pragma unroll 4
    for (int k0 = 0; k0 < H; k0 += 32) {
      v16bf a0 = load_frag_a(hsh, H, 0,  k0);
      v16bf a1 = load_frag_a(hsh, H, 16, k0);
      v16bf bb = load_frag_b(whh_sh, H, lrow0, k0);
      acc0 = __builtin_amdgcn_wmma_f32_16x16x32_bf16(false, a0, false, bb, (short)0, acc0, false, false);
      acc1 = __builtin_amdgcn_wmma_f32_16x16x32_bf16(false, a1, false, bb, (short)0, acc1, false, false);
    }

    // Add the precomputed input projection and park gates in LDS.
    {
      const int n  = lane & 15;
      const int mo = (lane >> 4) * 8;
      const int cl = wave * 16 + n;                 // local gate col (0..127) == q*32 + jb + n
      const int gc = q * H + hc0 + jb + n;          // global gate col in [0,4H)
      const float* xp = xproj + (size_t)t * BB * G + gc;
#pragma unroll
      for (int v = 0; v < 8; ++v) {
        const int r0 = mo + v, r1 = 16 + mo + v;
        gsh[r0 * GCOLS + cl] = acc0[v] + xp[(size_t)r0 * G];
        gsh[r1 * GCOLS + cl] = acc1[v] + xp[(size_t)r1 * G];
      }
    }
    __syncthreads();

    // Fused LSTM cell for our 32 h-columns; c-state persistent in LDS.
    uint16_t* hnext = hstage + (size_t)((t + 1) & 1) * BB * H;
    uint16_t* hout  = hs_out + (size_t)t * BB * H;
    for (int e = tid; e < BB * HSLICE; e += 256) {
      const int br = e >> 5, j = e & 31;
      const float iv = fast_sigmoid(gsh[br * GCOLS + j]);
      const float fv = fast_sigmoid(gsh[br * GCOLS + 32 + j]);
      const float gv = tanhf(gsh[br * GCOLS + 64 + j]);
      const float ov = fast_sigmoid(gsh[br * GCOLS + 96 + j]);
      const float cc = fv * csh[e] + iv * gv;
      csh[e] = cc;
      const uint16_t hb = f2bf(ov * tanhf(cc));
      hnext[(size_t)br * H + hc0 + j] = hb;
      hout [(size_t)br * H + hc0 + j] = hb;
    }

    grid_sync(cnt, (volatile int*)flag, t + 1);
  }
}

// ---------------- final projection + softmax (tiny) ----------------
__global__ void __launch_bounds__(256)
proj_softmax_k(const uint16_t* __restrict__ hs, const float* __restrict__ W3,
               const float* __restrict__ b3, float* __restrict__ out) {
  const int row = blockIdx.x * blockDim.x + threadIdx.x;   // 0..M1-1
  if (row >= M1) return;
  const uint16_t* hp = hs + (size_t)row * H;
  float acc[OUT];
#pragma unroll
  for (int o = 0; o < OUT; ++o) acc[o] = b3[o];
  for (int k = 0; k < H; ++k) {
    const float hv = bf2f(hp[k]);
    const float* wk = W3 + k;
#pragma unroll
    for (int o = 0; o < OUT; ++o) acc[o] += hv * wk[(size_t)o * H];
  }
  float m = acc[0];
#pragma unroll
  for (int o = 1; o < OUT; ++o) m = fmaxf(m, acc[o]);
  float s = 0.0f;
#pragma unroll
  for (int o = 0; o < OUT; ++o) { acc[o] = __expf(acc[o] - m); s += acc[o]; }
  const float inv = 1.0f / s;
  float* op = out + (size_t)row * OUT;
#pragma unroll
  for (int o = 0; o < OUT; ++o) op[o] = acc[o] * inv;
}

// ---------------- host orchestration ----------------
extern "C" void kernel_launch(void* const* d_in, const int* in_sizes, int n_in,
                              void* d_out, int out_size, void* d_ws, size_t ws_size,
                              hipStream_t stream) {
  const float* data  = (const float*)d_in[0];
  const float* W_ih1 = (const float*)d_in[1];
  const float* W_hh1 = (const float*)d_in[2];
  const float* b1    = (const float*)d_in[3];
  const float* W_ih2 = (const float*)d_in[4];
  const float* W_hh2 = (const float*)d_in[5];
  const float* b2    = (const float*)d_in[6];
  const float* W3    = (const float*)d_in[7];
  const float* b3    = (const float*)d_in[8];

  char* ws = (char*)d_ws;
  size_t off = 0;
  auto carve = [&](size_t bytes) -> void* {
    off = (off + 255) & ~(size_t)255;
    void* p = ws + off;
    off += bytes;
    return p;
  };

  float*    gates   = (float*)   carve((size_t)M1 * G * 4);     // reused by both layers
  uint16_t* hs1     = (uint16_t*)carve((size_t)M1 * H * 2);
  uint16_t* hs2     = (uint16_t*)carve((size_t)M1 * H * 2);
  uint16_t* data_bf = (uint16_t*)carve((size_t)M1 * I * 2);
  uint16_t* wih1_bf = (uint16_t*)carve((size_t)G * I * 2);
  uint16_t* whh1_bf = (uint16_t*)carve((size_t)G * H * 2);
  uint16_t* wih2_bf = (uint16_t*)carve((size_t)G * H * 2);
  uint16_t* whh2_bf = (uint16_t*)carve((size_t)G * H * 2);
  uint16_t* hstage  = (uint16_t*)carve((size_t)2 * BB * H * 2);
  int*      sync    = (int*)     carve(256);

  const size_t recur_lds = (size_t)(GCOLS + BB) * H * 2   // whh + h staging (bf16)
                         + (size_t)BB * GCOLS * 4         // gates f32
                         + (size_t)BB * HSLICE * 4;       // c-state f32

  auto conv = [&](const float* src, uint16_t* dst, int n) {
    int blocks = (n + 255) / 256; if (blocks > 2048) blocks = 2048;
    f32_to_bf16_k<<<blocks, 256, 0, stream>>>(src, dst, n);
  };
  conv(data,  data_bf, M1 * I);
  conv(W_ih1, wih1_bf, G * I);
  conv(W_hh1, whh1_bf, G * H);
  conv(W_ih2, wih2_bf, G * H);
  conv(W_hh2, whh2_bf, G * H);

  const int hdw = 2 * BB * H / 2;  // dwords in hstage
  init_state_k<<<(hdw + 255) / 256, 256, 0, stream>>>((uint32_t*)hstage, hdw, sync);

  // Layer 1: input projection (K=I), then recurrence.
  gemm_bias_wmma<<<(M1 / 32) * (G / 256), 256, 0, stream>>>(data_bf, wih1_bf, b1, gates, M1, G, I);
  lstm_recur_wmma<<<NWG, 256, recur_lds, stream>>>(gates, whh1_bf, hstage, hs1, &sync[0], &sync[1]);

  // Re-zero h staging + sync for layer 2.
  init_state_k<<<(hdw + 255) / 256, 256, 0, stream>>>((uint32_t*)hstage, hdw, sync);

  // Layer 2: input projection (K=H) from hs1, then recurrence.
  gemm_bias_wmma<<<(M1 / 32) * (G / 256), 256, 0, stream>>>(hs1, wih2_bf, b2, gates, M1, G, H);
  lstm_recur_wmma<<<NWG, 256, recur_lds, stream>>>(gates, whh2_bf, hstage, hs2, &sync[0], &sync[1]);

  // Head: logits + softmax.
  proj_softmax_k<<<(M1 + 255) / 256, 256, 0, stream>>>(hs2, W3, b3, (float*)d_out);
}
